// MoELayer_84181359001995
// MI455X (gfx1250) — matrix operand order repaired and checked by reading, not compiled, not run on hardware
//
#include <hip/hip_runtime.h>

#define N_TOK 4096
#define C_DIM 1024
#define F_DIM 4096
#define N_EXP 8
#define TOPK  2
#define EPSR  1e-6f

typedef __attribute__((ext_vector_type(16))) __bf16 v16bf;
typedef __attribute__((ext_vector_type(8)))  __bf16 v8bf;
typedef __attribute__((ext_vector_type(8)))  float  v8f;
typedef __attribute__((ext_vector_type(2)))  float  v2f;

// Build a v16bf WMMA fragment from two contiguous 8-element (16B) LDS runs.
__device__ __forceinline__ v16bf frag_from_row(const __bf16* row, int o0, int o1) {
    const v8bf a0 = *(const v8bf*)(row + o0);
    const v8bf a1 = *(const v8bf*)(row + o1);
    v16bf r;
#pragma unroll
    for (int i = 0; i < 8; ++i) { r[i] = a0[i]; r[i + 8] = a1[i]; }
    return r;
}

// CDNA5 async global->LDS DMA: one 16B transfer per lane, tracked by ASYNCcnt.
__device__ __forceinline__ void async_copy16(unsigned lds_addr, const void* gptr) {
    asm volatile("global_load_async_to_lds_b128 %0, %1, off"
                 :: "v"(lds_addr), "v"(gptr) : "memory");
}
__device__ __forceinline__ void async_wait0() {
    asm volatile("s_wait_asynccnt 0" ::: "memory");
}

__global__ void k_cvt_x(const float* __restrict__ x, __bf16* __restrict__ xb, int n) {
    int i = blockIdx.x * blockDim.x + threadIdx.x;
    if (i < n) xb[i] = (__bf16)x[i];
}

__global__ void k_zero(float* __restrict__ p, int n) {
    int i = blockIdx.x * blockDim.x + threadIdx.x;
    if (i < n) p[i] = 0.0f;
}

// ---------------------------------------------------------------------------
// Router projection: xp = X @ w_in  (fp32 WMMA 16x16x4, keeps routing exact-ish)
// Block tile 32x64, 8 waves (2x4), wave tile 16x16. A tile loaded async.
// ---------------------------------------------------------------------------
__global__ __launch_bounds__(256) void k_gemm_xp(const float* __restrict__ X,
                                                 const float* __restrict__ W,
                                                 float* __restrict__ XP) {
    __shared__ float As[32][32];   // [m][k]
    __shared__ float Bs[64][32];   // transposed: [n][k]
    const int tid  = threadIdx.x;
    const int row0 = blockIdx.x * 32;
    const int col0 = blockIdx.y * 64;
    const int wid  = tid >> 5, lane = tid & 31;
    const int wr   = wid >> 2, wc   = wid & 3;
    const int am   = wr * 16 + (lane & 15);          // A row (local)
    const int nl   = wc * 16 + (lane & 15);          // B/C col (local)
    const int kb   = (lane >> 4) * 2;                // K sub-offset per lane half
    const unsigned as_lds = (unsigned)(uintptr_t)&As[0][0] + tid * 16;
    const float* agp = X + (size_t)(row0 + (tid >> 3)) * C_DIM + (tid & 7) * 4;
    v8f acc = {};
    for (int kc = 0; kc < C_DIM; kc += 32) {
        async_copy16(as_lds, agp + kc);              // A: 4KB via async DMA
#pragma unroll
        for (int r = 0; r < 8; ++r) {                // B: transposed store
            int idx = r * 256 + tid;
            int bk = idx >> 6, bn = idx & 63;
            Bs[bn][bk] = W[(kc + bk) * C_DIM + col0 + bn];
        }
        async_wait0();
        __syncthreads();
#pragma unroll
        for (int s = 0; s < 8; ++s) {                // 8 K=4 steps per 32-chunk
            const int k0 = s * 4 + kb;
            v2f a; a.x = As[am][k0]; a.y = As[am][k0 + 1];
            v2f b; b.x = Bs[nl][k0]; b.y = Bs[nl][k0 + 1];
            acc = __builtin_amdgcn_wmma_f32_16x16x4_f32(false, a, false, b,
                                                        (short)0, acc, false, false);
        }
        __syncthreads();
    }
    const int srow = row0 + wr * 16 + (lane >> 4) * 8;
    const int scol = col0 + nl;
#pragma unroll
    for (int i = 0; i < 8; ++i)
        XP[(srow + i) * C_DIM + scol] = acc[i];
}

// ---------------------------------------------------------------------------
// Router: L2-normalize, label similarity, softmax, top-2, comb matrix.
// One wave32 per token.
// ---------------------------------------------------------------------------
__global__ __launch_bounds__(256) void k_route(const float* __restrict__ XP,
                                               const float* __restrict__ labels,
                                               float* __restrict__ comb,
                                               float* __restrict__ out_w,
                                               float* __restrict__ out_i,
                                               float* __restrict__ out_s) {
    const int wid = threadIdx.x >> 5, lane = threadIdx.x & 31;
    const int tok = blockIdx.x * 8 + wid;
    const float* xr = XP + (size_t)tok * C_DIM;
    float ss = 0.0f;
    float dot[N_EXP] = {};
    for (int c = lane; c < C_DIM; c += 32) {
        float v = xr[c];
        ss += v * v;
#pragma unroll
        for (int e = 0; e < N_EXP; ++e) dot[e] += v * labels[e * C_DIM + c];
    }
#pragma unroll
    for (int off = 16; off >= 1; off >>= 1) {
        ss += __shfl_xor(ss, off, 32);
#pragma unroll
        for (int e = 0; e < N_EXP; ++e) dot[e] += __shfl_xor(dot[e], off, 32);
    }
    if (lane == 0) {
        const float inv = 1.0f / (sqrtf(ss) + EPSR);
        float sc[N_EXP], mx = -1e30f;
#pragma unroll
        for (int e = 0; e < N_EXP; ++e) { sc[e] = dot[e] * inv; mx = fmaxf(mx, sc[e]); }
        float p[N_EXP], sum = 0.0f;
#pragma unroll
        for (int e = 0; e < N_EXP; ++e) { p[e] = expf(sc[e] - mx); sum += p[e]; }
        const float is = 1.0f / sum;
#pragma unroll
        for (int e = 0; e < N_EXP; ++e) p[e] *= is;
        int i0 = 0;
#pragma unroll
        for (int e = 1; e < N_EXP; ++e) if (p[e] > p[i0]) i0 = e;   // first idx on tie
        int i1 = (i0 == 0) ? 1 : 0;
#pragma unroll
        for (int e = 0; e < N_EXP; ++e) if (e != i0 && p[e] > p[i1]) i1 = e;
#pragma unroll
        for (int e = 0; e < N_EXP; ++e) {
            float w = (e == i0) ? p[i0] : ((e == i1) ? p[i1] : 0.0f);
            comb[tok * N_EXP + e]  = w;
            out_s[tok * N_EXP + e] = sc[e];
        }
        out_w[tok * TOPK + 0] = p[i0];
        out_w[tok * TOPK + 1] = p[i1];
        out_i[tok * TOPK + 0] = (float)i0;
        out_i[tok * TOPK + 1] = (float)i1;
    }
}

// ---------------------------------------------------------------------------
// Expert FFN stage 1: H = relu(X @ w_fc[e])^2, bf16 WMMA, skip inactive tiles.
// Block tile 64x128, 8 waves (2x4), wave tile 32x32 (4 WMMAs reuse A/B frags).
// A tile (bf16, no conversion) loaded with async global->LDS DMA.
// ---------------------------------------------------------------------------
__global__ __launch_bounds__(256) void k_ffn1(const __bf16* __restrict__ Xb,
                                              const float* __restrict__ Wfc,
                                              const float* __restrict__ comb, int e,
                                              __bf16* __restrict__ H) {
    __shared__ __bf16 As[64][32];     // [m][k]
    __shared__ __bf16 Bs[128][32];    // transposed: [n][k]
    __shared__ int s_active;
    const int tid  = threadIdx.x;
    const int row0 = blockIdx.x * 64;
    const int col0 = blockIdx.y * 128;
    if (tid == 0) {
        int act = 0;
        for (int r = 0; r < 64; ++r) act |= (comb[(row0 + r) * N_EXP + e] != 0.0f);
        s_active = act;
    }
    __syncthreads();
    if (!s_active) return;            // top-2 sparsity: skip unrouted token tiles

    const int wid  = tid >> 5, lane = tid & 31;
    const int wr   = wid >> 2, wc   = wid & 3;
    const int am0  = wr * 32 + (lane & 15);
    const int bn0  = wc * 32 + (lane & 15);
    const int half = lane >> 4;
    const int kb   = (lane >> 4) * 16;
    const unsigned as_lds = (unsigned)(uintptr_t)&As[0][0] + tid * 16;
    const __bf16* agp = Xb + (size_t)(row0 + (tid >> 2)) * C_DIM + (tid & 3) * 8;
    v8f acc00 = {}, acc01 = {}, acc10 = {}, acc11 = {};
    for (int kc = 0; kc < C_DIM; kc += 32) {
        async_copy16(as_lds, agp + kc);              // A: 4KB via async DMA
#pragma unroll
        for (int r = 0; r < 16; ++r) {               // B: cvt fp32->bf16, transpose
            int idx = r * 256 + tid;
            int bk = idx >> 7, bn = idx & 127;
            Bs[bn][bk] = (__bf16)Wfc[(kc + bk) * F_DIM + col0 + bn];
        }
        if (kc + 32 < C_DIM)                         // hint next B chunk into L2
            __builtin_prefetch(&Wfc[(kc + 32 + (tid >> 7)) * F_DIM + col0 + (tid & 127)], 0, 0);
        async_wait0();
        __syncthreads();
        v16bf a0 = frag_from_row(&As[am0][0],      kb, kb + 8);
        v16bf a1 = frag_from_row(&As[am0 + 16][0], kb, kb + 8);
        v16bf b0 = frag_from_row(&Bs[bn0][0],      kb, kb + 8);
        v16bf b1 = frag_from_row(&Bs[bn0 + 16][0], kb, kb + 8);
        acc00 = __builtin_amdgcn_wmma_f32_16x16x32_bf16(false, a0, false, b0, (short)0, acc00, false, false);
        acc01 = __builtin_amdgcn_wmma_f32_16x16x32_bf16(false, a0, false, b1, (short)0, acc01, false, false);
        acc10 = __builtin_amdgcn_wmma_f32_16x16x32_bf16(false, a1, false, b0, (short)0, acc10, false, false);
        acc11 = __builtin_amdgcn_wmma_f32_16x16x32_bf16(false, a1, false, b1, (short)0, acc11, false, false);
        __syncthreads();
    }
    const int srow0 = row0 + wr * 32 + (lane >> 4) * 8;
    const int scol0 = col0 + wc * 32 + (lane & 15);
#pragma unroll
    for (int i = 0; i < 8; ++i) {
        float h;
        h = fmaxf(acc00[i], 0.0f); H[(srow0 + i) * F_DIM + scol0]           = (__bf16)(h * h);
        h = fmaxf(acc01[i], 0.0f); H[(srow0 + i) * F_DIM + scol0 + 16]      = (__bf16)(h * h);
        h = fmaxf(acc10[i], 0.0f); H[(srow0 + 16 + i) * F_DIM + scol0]      = (__bf16)(h * h);
        h = fmaxf(acc11[i], 0.0f); H[(srow0 + 16 + i) * F_DIM + scol0 + 16] = (__bf16)(h * h);
    }
}

// ---------------------------------------------------------------------------
// Expert FFN stage 2: out += comb[:,e] * (H @ w_out[e]), same tiling/skip.
// ---------------------------------------------------------------------------
__global__ __launch_bounds__(256) void k_ffn2(const __bf16* __restrict__ H,
                                              const float* __restrict__ Wout,
                                              const float* __restrict__ comb, int e,
                                              float* __restrict__ out) {
    __shared__ __bf16 As[64][32];
    __shared__ __bf16 Bs[128][32];
    __shared__ float  s_comb[64];
    __shared__ int    s_active;
    const int tid  = threadIdx.x;
    const int row0 = blockIdx.x * 64;
    const int col0 = blockIdx.y * 128;
    if (tid < 64) s_comb[tid] = comb[(row0 + tid) * N_EXP + e];
    __syncthreads();
    if (tid == 0) {
        int act = 0;
        for (int r = 0; r < 64; ++r) act |= (s_comb[r] != 0.0f);
        s_active = act;
    }
    __syncthreads();
    if (!s_active) return;

    const int wid  = tid >> 5, lane = tid & 31;
    const int wr   = wid >> 2, wc   = wid & 3;
    const int am0  = wr * 32 + (lane & 15);
    const int bn0  = wc * 32 + (lane & 15);
    const int half = lane >> 4;
    const int kb   = (lane >> 4) * 16;
    const unsigned as_lds = (unsigned)(uintptr_t)&As[0][0] + tid * 16;
    const __bf16* agp = H + (size_t)(row0 + (tid >> 2)) * F_DIM + (tid & 3) * 8;
    v8f acc00 = {}, acc01 = {}, acc10 = {}, acc11 = {};
    for (int kc = 0; kc < F_DIM; kc += 32) {
        async_copy16(as_lds, agp + kc);
#pragma unroll
        for (int r = 0; r < 16; ++r) {
            int idx = r * 256 + tid;
            int bk = idx >> 7, bn = idx & 127;
            Bs[bn][bk] = (__bf16)Wout[(kc + bk) * C_DIM + col0 + bn];
        }
        if (kc + 32 < F_DIM)
            __builtin_prefetch(&Wout[(kc + 32 + (tid >> 7)) * C_DIM + col0 + (tid & 127)], 0, 0);
        async_wait0();
        __syncthreads();
        v16bf a0 = frag_from_row(&As[am0][0],      kb, kb + 8);
        v16bf a1 = frag_from_row(&As[am0 + 16][0], kb, kb + 8);
        v16bf b0 = frag_from_row(&Bs[bn0][0],      kb, kb + 8);
        v16bf b1 = frag_from_row(&Bs[bn0 + 16][0], kb, kb + 8);
        acc00 = __builtin_amdgcn_wmma_f32_16x16x32_bf16(false, a0, false, b0, (short)0, acc00, false, false);
        acc01 = __builtin_amdgcn_wmma_f32_16x16x32_bf16(false, a0, false, b1, (short)0, acc01, false, false);
        acc10 = __builtin_amdgcn_wmma_f32_16x16x32_bf16(false, a1, false, b0, (short)0, acc10, false, false);
        acc11 = __builtin_amdgcn_wmma_f32_16x16x32_bf16(false, a1, false, b1, (short)0, acc11, false, false);
        __syncthreads();
    }
    const int lrow0 = wr * 32 + (lane >> 4) * 8;
    const int scol0 = col0 + wc * 32 + (lane & 15);
#pragma unroll
    for (int i = 0; i < 8; ++i) {
        const float w0 = s_comb[lrow0 + i];
        const float w1 = s_comb[lrow0 + 16 + i];
        const int r0 = (row0 + lrow0 + i) * C_DIM;
        const int r1 = (row0 + lrow0 + 16 + i) * C_DIM;
        out[r0 + scol0]      += w0 * acc00[i];
        out[r0 + scol0 + 16] += w0 * acc01[i];
        out[r1 + scol0]      += w1 * acc10[i];
        out[r1 + scol0 + 16] += w1 * acc11[i];
    }
}

extern "C" void kernel_launch(void* const* d_in, const int* in_sizes, int n_in,
                              void* d_out, int out_size, void* d_ws, size_t ws_size,
                              hipStream_t stream) {
    (void)in_sizes; (void)n_in; (void)out_size; (void)ws_size;
    const float* x      = (const float*)d_in[0];
    const float* w_in   = (const float*)d_in[1];
    const float* labels = (const float*)d_in[2];
    const float* w_fc   = (const float*)d_in[3];
    const float* w_out  = (const float*)d_in[4];

    float* out   = (float*)d_out;                       // [N, C]
    float* out_w = out + (size_t)N_TOK * C_DIM;         // [N, 2]
    float* out_i = out_w + (size_t)N_TOK * TOPK;        // [N, 2] (as float)
    float* out_s = out_i + (size_t)N_TOK * TOPK;        // [N, 8]

    char*   ws   = (char*)d_ws;
    float*  xp   = (float*)(ws);                        // N*C fp32      (16.0 MB)
    float*  comb = (float*)(ws + 16777216);             // N*E fp32      (0.13 MB)
    __bf16* Xb   = (__bf16*)(ws + 16908288);            // N*C bf16      ( 8.0 MB)
    __bf16* H    = (__bf16*)(ws + 25296896);            // N*F bf16      (32.0 MB)

    k_cvt_x<<<(N_TOK * C_DIM + 255) / 256, 256, 0, stream>>>(x, Xb, N_TOK * C_DIM);
    k_zero <<<(N_TOK * C_DIM + 255) / 256, 256, 0, stream>>>(out, N_TOK * C_DIM);

    dim3 g1(N_TOK / 32, C_DIM / 64);
    k_gemm_xp<<<g1, 256, 0, stream>>>(x, w_in, xp);
    k_route<<<N_TOK / 8, 256, 0, stream>>>(xp, labels, comb, out_w, out_i, out_s);

    for (int e = 0; e < N_EXP; ++e) {
        dim3 gf1(N_TOK / 64, F_DIM / 128);
        k_ffn1<<<gf1, 256, 0, stream>>>(Xb, w_fc + (size_t)e * C_DIM * F_DIM, comb, e, H);
        dim3 gf2(N_TOK / 64, C_DIM / 128);
        k_ffn2<<<gf2, 256, 0, stream>>>(H, w_out + (size_t)e * F_DIM * C_DIM, comb, e, out);
    }
}
